// QNet_61134564491754
// MI455X (gfx1250) — matrix operand order repaired and checked by reading, not compiled, not run on hardware
//
#include <hip/hip_runtime.h>

#define B_    4096
#define N_    16
#define HX_   32
#define NOBS_ 150
#define NACT_ 10
#define OBSD_ 160

typedef __attribute__((ext_vector_type(16))) _Float16 v16h;
typedef __attribute__((ext_vector_type(8)))  float    v8f;

#define DEV static __device__ __forceinline__

DEV v8f wmma16(v16h a, v16h b, v8f c) {
  // D = A(16x32 f16) x B(32x16 f16) + C(16x16 f32)
  return __builtin_amdgcn_wmma_f32_16x16x32_f16(
      /*neg_a=*/false, a, /*neg_b=*/false, b,
      /*c_mod=*/(short)0, c, /*reuse_a=*/false, /*reuse_b=*/false);
}

DEV float sigmoidf_(float x) { return 1.f / (1.f + __expf(-x)); }

// ---- A fragment (16x32, f16) built from f32 source, row-major, zero-pad K.
// Lane layout (ISA 7.12.2): lanes 0-15 row=lane, K = kb+{0..7, 16..23};
// lanes 16-31 row=lane-16, K = kb+{8..15, 24..31}.
DEV v16h a_frag_f32(const float* base, int row_stride, int kbase, int kmax,
                    int lane, bool* anz) {
  const int row  = lane & 15;
  const int koff = (lane >> 4) << 3;            // 0 or 8
  const float* p = base + (size_t)row * row_stride + kbase + koff;
  v16h a;
  if (kbase + 32 <= kmax) {
#pragma unroll
    for (int k = 0; k < 8; ++k) {
      float v0 = p[k], v1 = p[16 + k];
      if (anz) *anz |= (v0 != 0.f) || (v1 != 0.f);
      a[k] = (_Float16)v0; a[8 + k] = (_Float16)v1;
    }
  } else {
#pragma unroll
    for (int k = 0; k < 8; ++k) {
      int k0 = kbase + koff + k, k1 = k0 + 16;
      float v0 = (k0 < kmax) ? p[k]      : 0.f;
      float v1 = (k1 < kmax) ? p[16 + k] : 0.f;
      if (anz) *anz |= (v0 != 0.f) || (v1 != 0.f);
      a[k] = (_Float16)v0; a[8 + k] = (_Float16)v1;
    }
  }
  return a;
}

// ---- A fragment from LDS f16 tile (row-major, stride in halves)
DEV v16h a_frag_lds(const _Float16* lds, int stride, int kbase, int lane) {
  const int row  = lane & 15;
  const int koff = (lane >> 4) << 3;
  const _Float16* p = lds + row * stride + kbase + koff;
  v16h a;
#pragma unroll
  for (int k = 0; k < 8; ++k) { a[k] = p[k]; a[8 + k] = p[16 + k]; }
  return a;
}

// ---- B fragment (32x16 f16) from LDS weights stored [out][in] row-major.
// Lanes 0-15: col=lane, K = kb+{0..7, 16..23}; lanes 16-31: col=lane-16, K = kb+{8..15, 24..31}.
DEV v16h b_frag_lds(const _Float16* lds, int stride, int nbase, int kbase, int lane) {
  const int col = nbase + (lane & 15);
  const int k0  = kbase + ((lane >> 4) << 3);   // 0 or 8
  const _Float16* p = lds + col * stride + k0;
  v16h b;
#pragma unroll
  for (int k = 0; k < 8; ++k) { b[k] = p[k]; b[8 + k] = p[16 + k]; }
  return b;
}

// ---- C-layout (8 f32/lane) -> LDS f16 tile with bias+relu
DEV void c_to_lds_relu(_Float16* lds, int stride, int nbase, v8f acc,
                       float bias, int lane) {
  const int col = nbase + (lane & 15);
  const int rb  = (lane >> 4) << 3;
#pragma unroll
  for (int r = 0; r < 8; ++r) {
    float v = acc[r] + bias;
    v = v > 0.f ? v : 0.f;
    lds[(rb + r) * stride + col] = (_Float16)v;
  }
}

// ---- stage f32 global weights -> f16 LDS, zero-padding K from kin to kpad
DEV void stage_w(_Float16* dst, const float* src, int rows, int kin, int kpad,
                 int tid, int nthr) {
  for (int idx = tid; idx < rows * kpad; idx += nthr) {
    int r = idx / kpad, k = idx - r * kpad;
    dst[idx] = (_Float16)((k < kin) ? src[(size_t)r * kin + k] : 0.f);
  }
}

// ---- async DMA of one contiguous 16x150-f32 comm tile (9600 B, 16B aligned)
// into LDS: 600 x b128 chunks, lanes take consecutive chunks (fully coalesced).
// Issues exactly 19 GLOBAL_LOAD_ASYNC_TO_LDS_B128 per wave (ASYNCcnt += 19).
#define TILE_BYTES   (16 * NOBS_ * 4)          // 9600
#define TILE_CHUNKS  (TILE_BYTES / 16)         // 600
#define ASYNC_OPS_PER_TILE ((TILE_CHUNKS + 31) / 32)   // 19

DEV void stage_comm_async(const float* gsrc, const float* lbuf, int lane) {
  unsigned long long ga = (unsigned long long)(uintptr_t)gsrc +
                          (unsigned long long)lane * 16u;
  unsigned lo = (unsigned)(uintptr_t)lbuf + (unsigned)lane * 16u;  // LDS offset
  for (int idx = lane; idx < TILE_CHUNKS; idx += 32) {
    asm volatile("global_load_async_to_lds_b128 %0, %1, off"
                 :: "v"(lo), "v"(ga) : "memory");
    ga += 32ull * 16ull;
    lo += 32u * 16u;
  }
}

// ============================================================================
// Kernel 0: zero n_comm + any-flags
// ============================================================================
__global__ void qnet_init_kernel(float* ncomm, unsigned* flags) {
  int idx = blockIdx.x * blockDim.x + threadIdx.x;
  if (idx < B_ * N_) ncomm[idx] = 0.f;
  if (idx < N_) flags[idx] = 0u;
}

// ============================================================================
// Kernel 1: self path. Per (agent i, 16 b-rows) tile:
//   feat = relu(relu(obs*fw1+fb1)*fw2+fb2); x_gru = GRU(feat, hidden);
//   gh_c = hidden*cWhh + cbhh   (precomputed for the comm kernel)
// grid(16,16), block 128 (4 waves); each wave does 4 tiles of 16 rows.
// ============================================================================
__global__ __launch_bounds__(128) void qnet_self_kernel(
    const float* __restrict__ obs, const float* __restrict__ hidden,
    const float* __restrict__ fw1, const float* __restrict__ fb1,
    const float* __restrict__ fw2, const float* __restrict__ fb2,
    const float* __restrict__ gWih, const float* __restrict__ gWhh,
    const float* __restrict__ gbih, const float* __restrict__ gbhh,
    const float* __restrict__ cWhh, const float* __restrict__ cbhh,
    float* __restrict__ feat_out, float* __restrict__ xgru_out,
    float* __restrict__ ghc_out) {
  const int i    = blockIdx.y;
  const int tid  = threadIdx.x;
  const int lane = tid & 31;
  const int wave = tid >> 5;

  __shared__ _Float16 s_fw1[64 * OBSD_];
  __shared__ _Float16 s_fw2[32 * 64];
  __shared__ _Float16 s_gWih[96 * 32];
  __shared__ _Float16 s_gWhh[96 * 32];
  __shared__ _Float16 s_cWhh[96 * 32];
  __shared__ _Float16 s_h1[4][16 * 64];
  __shared__ _Float16 s_ft[4][16 * 32];

  stage_w(s_fw1, fw1 + (size_t)i * 64 * OBSD_, 64, OBSD_, OBSD_, tid, 128);
  stage_w(s_fw2, fw2 + (size_t)i * 32 * 64, 32, 64, 64, tid, 128);
  stage_w(s_gWih, gWih + (size_t)i * 96 * 32, 96, 32, 32, tid, 128);
  stage_w(s_gWhh, gWhh + (size_t)i * 96 * 32, 96, 32, 32, tid, 128);
  stage_w(s_cWhh, cWhh + (size_t)i * 96 * 32, 96, 32, 32, tid, 128);
  __syncthreads();

  const int c  = lane & 15;
  const int rb = (lane >> 4) << 3;

  for (int t = 0; t < 4; ++t) {
    const int b0 = blockIdx.x * 256 + wave * 64 + t * 16;   // rows b0..b0+15
    const float* obase = obs + ((size_t)b0 * N_ + i) * OBSD_;
    const float* hbase = hidden + ((size_t)b0 * N_ + i) * HX_;

    // ---- layer 1: obs(160) -> 64
    v16h A1[5];
#pragma unroll
    for (int kb = 0; kb < 5; ++kb)
      A1[kb] = a_frag_f32(obase, N_ * OBSD_, kb * 32, OBSD_, lane, nullptr);
#pragma unroll
    for (int nt = 0; nt < 4; ++nt) {
      v8f acc = {};
#pragma unroll
      for (int kb = 0; kb < 5; ++kb)
        acc = wmma16(A1[kb], b_frag_lds(s_fw1, OBSD_, nt * 16, kb * 32, lane), acc);
      c_to_lds_relu(s_h1[wave], 64, nt * 16, acc, fb1[i * 64 + nt * 16 + c], lane);
    }

    // ---- layer 2: 64 -> 32 (feat)
    v16h A2[2];
#pragma unroll
    for (int kb = 0; kb < 2; ++kb) A2[kb] = a_frag_lds(s_h1[wave], 64, kb * 32, lane);
#pragma unroll
    for (int nt = 0; nt < 2; ++nt) {
      v8f acc = {};
#pragma unroll
      for (int kb = 0; kb < 2; ++kb)
        acc = wmma16(A2[kb], b_frag_lds(s_fw2, 64, nt * 16, kb * 32, lane), acc);
      float bias = fb2[i * 32 + nt * 16 + c];
      c_to_lds_relu(s_ft[wave], 32, nt * 16, acc, bias, lane);
#pragma unroll
      for (int r = 0; r < 8; ++r) {
        float v = acc[r] + bias; v = v > 0.f ? v : 0.f;
        feat_out[((size_t)(b0 + rb + r) * N_ + i) * HX_ + nt * 16 + c] = v;
      }
    }

    // ---- GRU inputs as A-fragments
    v16h Af = a_frag_lds(s_ft[wave], 32, 0, lane);
    v16h Ah = a_frag_f32(hbase, N_ * HX_, 0, HX_, lane, nullptr);

    // ---- self GRU (two 16-col halves)
#pragma unroll
    for (int tt = 0; tt < 2; ++tt) {
      v8f gr = wmma16(Af, b_frag_lds(s_gWih, 32, (0 + tt) * 16, 0, lane), (v8f){});
      v8f hr = wmma16(Ah, b_frag_lds(s_gWhh, 32, (0 + tt) * 16, 0, lane), (v8f){});
      v8f gz = wmma16(Af, b_frag_lds(s_gWih, 32, (2 + tt) * 16, 0, lane), (v8f){});
      v8f hz = wmma16(Ah, b_frag_lds(s_gWhh, 32, (2 + tt) * 16, 0, lane), (v8f){});
      v8f gn = wmma16(Af, b_frag_lds(s_gWih, 32, (4 + tt) * 16, 0, lane), (v8f){});
      v8f hn = wmma16(Ah, b_frag_lds(s_gWhh, 32, (4 + tt) * 16, 0, lane), (v8f){});
      float bir = gbih[i * 96 + (0 + tt) * 16 + c], bhr = gbhh[i * 96 + (0 + tt) * 16 + c];
      float biz = gbih[i * 96 + (2 + tt) * 16 + c], bhz = gbhh[i * 96 + (2 + tt) * 16 + c];
      float bin = gbih[i * 96 + (4 + tt) * 16 + c], bhn = gbhh[i * 96 + (4 + tt) * 16 + c];
#pragma unroll
      for (int r = 0; r < 8; ++r) {
        size_t row = (size_t)(b0 + rb + r) * N_ + i;
        float rr = sigmoidf_(gr[r] + bir + hr[r] + bhr);
        float zz = sigmoidf_(gz[r] + biz + hz[r] + bhz);
        float nn = tanhf(gn[r] + bin + rr * (hn[r] + bhn));
        float hv = hidden[row * HX_ + tt * 16 + c];
        xgru_out[row * HX_ + tt * 16 + c] = (1.f - zz) * nn + zz * hv;
      }
    }

    // ---- gh_c = hidden*cWhh + cbhh (all 6 gate tiles)
#pragma unroll
    for (int g = 0; g < 6; ++g) {
      v8f acc = wmma16(Ah, b_frag_lds(s_cWhh, 32, g * 16, 0, lane), (v8f){});
      float bias = cbhh[i * 96 + g * 16 + c];
#pragma unroll
      for (int r = 0; r < 8; ++r)
        ghc_out[((size_t)(b0 + rb + r) * N_ + i) * 96 + g * 16 + c] = acc[r] + bias;
    }
  }
}

// ============================================================================
// Kernel 2: comm path (single pass over the 630 MB comm tensor).
// One tile = fixed (b, i), rows = partners j=0..15, so the sum over j
// completes inside the tile (no atomics for masked_sum).
// Tiles are DMA'd to LDS with GLOBAL_LOAD_ASYNC_TO_LDS_B128 (double-buffered
// per wave, ASYNCcnt-synchronized) so global traffic is fully coalesced.
// grid(64,16), block 128; each wave processes 16 tiles.
// ============================================================================
__global__ __launch_bounds__(128) void qnet_comm_kernel(
    const float* __restrict__ comm, const float* __restrict__ hidden,
    const float* __restrict__ cw1, const float* __restrict__ cb1,
    const float* __restrict__ cw2, const float* __restrict__ cb2,
    const float* __restrict__ cWih, const float* __restrict__ cbih,
    const float* __restrict__ ghc, float* __restrict__ msum,
    float* __restrict__ ncomm, unsigned* __restrict__ flags) {
  const int i    = blockIdx.y;
  const int tid  = threadIdx.x;
  const int lane = tid & 31;
  const int wave = tid >> 5;

  __shared__ _Float16 s_cw1[64 * OBSD_];   // 150 padded to 160
  __shared__ _Float16 s_cw2[32 * 64];
  __shared__ _Float16 s_cWih[96 * 32];
  __shared__ _Float16 s_c1[4][16 * 64];
  __shared__ _Float16 s_c2[4][16 * 32];
  __shared__ float    s_comm[4][2][16 * NOBS_];   // double-buffered DMA target

  stage_w(s_cw1, cw1 + (size_t)i * 64 * NOBS_, 64, NOBS_, OBSD_, tid, 128);
  stage_w(s_cw2, cw2 + (size_t)i * 32 * 64, 32, 64, 64, tid, 128);
  stage_w(s_cWih, cWih + (size_t)i * 96 * 32, 96, 32, 32, tid, 128);
  __syncthreads();

  const int c  = lane & 15;
  const int rb = (lane >> 4) << 3;
  const int bbase = blockIdx.x * 64 + wave * 16;

  // prefetch tile 0
  stage_comm_async(comm + (size_t)(((bbase + 0) * N_ + i) * N_) * NOBS_,
                   s_comm[wave][0], lane);

  for (int t = 0; t < 16; ++t) {
    const int b = bbase + t;
    if (t + 1 < 16) {
      stage_comm_async(comm + (size_t)(((b + 1) * N_ + i) * N_) * NOBS_,
                       s_comm[wave][(t + 1) & 1], lane);
      // one tile's worth (19 ops) may remain outstanding
      asm volatile("s_wait_asynccnt 19" ::: "memory");
    } else {
      asm volatile("s_wait_asynccnt 0" ::: "memory");
    }
    const float* cl = s_comm[wave][t & 1];   // resident tile, stride NOBS_

    // ---- build A (16 j-rows x 150->160) + per-row nonzero mask
    bool anz = false;
    v16h A1[5];
#pragma unroll
    for (int kb = 0; kb < 5; ++kb)
      A1[kb] = a_frag_f32(cl, NOBS_, kb * 32, NOBS_, lane, &anz);
    unsigned long long bal = __ballot(anz);
    unsigned rm = (unsigned)(bal | (bal >> 16)) & 0xFFFFu;  // bit j = mask[b,i,j]

    if (lane < 16) {
      if ((rm >> lane) & 1u) {
        atomicAdd(&ncomm[(size_t)b * N_ + lane], 1.f);   // n_comm sums over i
        atomicOr(&flags[lane], 1u);                      // any over (b,i)
      }
    }

    // ---- layer 1: 150 -> 64
#pragma unroll
    for (int nt = 0; nt < 4; ++nt) {
      v8f acc = {};
#pragma unroll
      for (int kb = 0; kb < 5; ++kb)
        acc = wmma16(A1[kb], b_frag_lds(s_cw1, OBSD_, nt * 16, kb * 32, lane), acc);
      c_to_lds_relu(s_c1[wave], 64, nt * 16, acc, cb1[i * 64 + nt * 16 + c], lane);
    }

    // ---- layer 2: 64 -> 32
    v16h A2[2];
#pragma unroll
    for (int kb = 0; kb < 2; ++kb) A2[kb] = a_frag_lds(s_c1[wave], 64, kb * 32, lane);
#pragma unroll
    for (int nt = 0; nt < 2; ++nt) {
      v8f acc = {};
#pragma unroll
      for (int kb = 0; kb < 2; ++kb)
        acc = wmma16(A2[kb], b_frag_lds(s_cw2, 64, nt * 16, kb * 32, lane), acc);
      c_to_lds_relu(s_c2[wave], 32, nt * 16, acc, cb2[i * 32 + nt * 16 + c], lane);
    }

    // ---- layer 3 (gi_c) + comm GRU + masked sum over j
    v16h A3 = a_frag_lds(s_c2[wave], 32, 0, lane);
    const float* gb = ghc + ((size_t)b * N_ + i) * 96;   // depends on (b,i) only
#pragma unroll
    for (int tt = 0; tt < 2; ++tt) {
      v8f gr = wmma16(A3, b_frag_lds(s_cWih, 32, (0 + tt) * 16, 0, lane), (v8f){});
      v8f gz = wmma16(A3, b_frag_lds(s_cWih, 32, (2 + tt) * 16, 0, lane), (v8f){});
      v8f gn = wmma16(A3, b_frag_lds(s_cWih, 32, (4 + tt) * 16, 0, lane), (v8f){});
      float ghr = gb[(0 + tt) * 16 + c], bir = cbih[i * 96 + (0 + tt) * 16 + c];
      float ghz = gb[(2 + tt) * 16 + c], biz = cbih[i * 96 + (2 + tt) * 16 + c];
      float ghn = gb[(4 + tt) * 16 + c], bin = cbih[i * 96 + (4 + tt) * 16 + c];
      float hv  = hidden[((size_t)b * N_ + i) * HX_ + tt * 16 + c];
      float psum = 0.f;
#pragma unroll
      for (int r = 0; r < 8; ++r) {
        float rr = sigmoidf_(gr[r] + bir + ghr);
        float zz = sigmoidf_(gz[r] + biz + ghz);
        float nn = tanhf(gn[r] + bin + rr * ghn);
        float x  = (1.f - zz) * nn + zz * hv;
        float m  = ((rm >> (rb + r)) & 1u) ? 1.f : 0.f;
        psum += m * x;
      }
      psum += __shfl_xor(psum, 16, 32);   // combine j-rows 0-7 with 8-15
      if (lane < 16)
        msum[((size_t)b * N_ + i) * HX_ + tt * 16 + lane] = psum;
    }
  }
}

// ============================================================================
// Kernel 3: branch select + denom + Q head
// ============================================================================
__global__ __launch_bounds__(256) void qnet_head_kernel(
    const float* __restrict__ feat, const float* __restrict__ xgru,
    const float* __restrict__ msum, const float* __restrict__ ncomm,
    const unsigned* __restrict__ flags,
    const float* __restrict__ qw, const float* __restrict__ qb,
    float* __restrict__ outq, float* __restrict__ outh) {
  int idx = blockIdx.x * blockDim.x + threadIdx.x;   // (b,n) row
  if (idx >= B_ * N_) return;
  const int n = idx & (N_ - 1);
  const bool any = flags[n] != 0u;
  float ncm = ncomm[idx];
  float denom = (ncm == 0.f) ? 1.f : ncm;

  float x[HX_], cv[HX_];
#pragma unroll
  for (int h = 0; h < HX_; ++h) {
    size_t o = (size_t)idx * HX_ + h;
    x[h]  = any ? xgru[o] : feat[o];
    cv[h] = any ? msum[o] / denom : 0.f;
    outh[o] = x[h];
  }
#pragma unroll
  for (int a = 0; a < NACT_; ++a) {
    const float* w = qw + ((size_t)n * NACT_ + a) * (2 * HX_);
    float acc = qb[n * NACT_ + a];
#pragma unroll
    for (int h = 0; h < HX_; ++h) acc += x[h] * w[h] + cv[h] * w[HX_ + h];
    outq[(size_t)idx * NACT_ + a] = acc;
  }
}

// ============================================================================
extern "C" void kernel_launch(void* const* d_in, const int* in_sizes, int n_in,
                              void* d_out, int out_size, void* d_ws, size_t ws_size,
                              hipStream_t stream) {
  (void)in_sizes; (void)n_in; (void)out_size; (void)ws_size;
  const float* obs    = (const float*)d_in[0];
  const float* hidden = (const float*)d_in[1];
  const float* comm   = (const float*)d_in[2];
  const float* fw1    = (const float*)d_in[3];
  const float* fb1    = (const float*)d_in[4];
  const float* fw2    = (const float*)d_in[5];
  const float* fb2    = (const float*)d_in[6];
  const float* cw1    = (const float*)d_in[7];
  const float* cb1    = (const float*)d_in[8];
  const float* cw2    = (const float*)d_in[9];
  const float* cb2    = (const float*)d_in[10];
  const float* gWih   = (const float*)d_in[11];
  const float* gWhh   = (const float*)d_in[12];
  const float* gbih   = (const float*)d_in[13];
  const float* gbhh   = (const float*)d_in[14];
  const float* cWih   = (const float*)d_in[15];
  const float* cWhh   = (const float*)d_in[16];
  const float* cbih   = (const float*)d_in[17];
  const float* cbhh   = (const float*)d_in[18];
  const float* qw     = (const float*)d_in[19];
  const float* qb     = (const float*)d_in[20];

  // workspace layout (floats)
  float* ws    = (float*)d_ws;
  float* ghc   = ws;                                  // B*N*96
  float* feat  = ghc  + (size_t)B_ * N_ * 96;         // B*N*32
  float* xgru  = feat + (size_t)B_ * N_ * HX_;        // B*N*32
  float* msum  = xgru + (size_t)B_ * N_ * HX_;        // B*N*32
  float* ncomm = msum + (size_t)B_ * N_ * HX_;        // B*N
  unsigned* flags = (unsigned*)(ncomm + (size_t)B_ * N_);  // N

  float* outq = (float*)d_out;                        // (B,N,NACT)
  float* outh = outq + (size_t)B_ * N_ * NACT_;       // (B,N,HX)

  qnet_init_kernel<<<dim3((B_ * N_ + 255) / 256), dim3(256), 0, stream>>>(ncomm, flags);

  qnet_self_kernel<<<dim3(B_ / 256, N_), dim3(128), 0, stream>>>(
      obs, hidden, fw1, fb1, fw2, fb2, gWih, gWhh, gbih, gbhh, cWhh, cbhh,
      feat, xgru, ghc);

  qnet_comm_kernel<<<dim3(B_ / 64, N_), dim3(128), 0, stream>>>(
      comm, hidden, cw1, cb1, cw2, cb2, cWih, cbih, ghc, msum, ncomm, flags);

  qnet_head_kernel<<<dim3((B_ * N_ + 255) / 256), dim3(256), 0, stream>>>(
      feat, xgru, msum, ncomm, flags, qw, qb, outq, outh);
}
